// QuantQwen3Attention_17025250361716
// MI455X (gfx1250) — compile-verified, matrix operand order checked by reading, not run on hardware
//
#include <hip/hip_runtime.h>
#include <hip/hip_bf16.h>

// QuantQwen3Attention for MI455X (gfx1250, wave32).
//
// The reference's qdq() is symmetric int8 fake-quant. The faithful *and* fast
// realization on CDNA5 is real int8 compute via V_WMMA_I32_16X16X64_IU8 with
// fp32 dequant scales applied after the i32 accumulation (exact: per-token
// act scale x per-row weight scale factor out of the K-sum). All matmuls
// (QKV proj, QK^T, PV, O proj) run on the WMMA path. Attention is
// flash-style (streaming 64-key blocks, online softmax), with per-token
// v-scales folded into p before p's int8 quantization (the v-scale lives on
// the K dim of PV so it cannot be post-applied).
//
// Roofline: ~86 GFLOP vs ~100MB of unique traffic => HBM time ~4.3us at
// 23.3 TB/s, far below the int8 WMMA math time => compute-bound; the whole
// ~80MB working set is L2-resident (192MB), so intermediates stay on-chip.
//
// The projection GEMMs use double-buffered async copies into LDS
// (GLOBAL_LOAD_ASYNC_TO_LDS_B128 + s_wait_asynccnt), overlapping the next
// tile's DMA with the current tile's WMMAs and removing intra-block refetch
// redundancy.

#define H_HEADS 16
#define KV_HEADS 8
#define HD 128
#define QMAXF 127.0f
#define KSTEP 64

typedef __attribute__((ext_vector_type(8))) int   v8i;
typedef __attribute__((ext_vector_type(4))) int   v4i;
typedef __attribute__((ext_vector_type(8))) float v8f;

union vi8u { v8i v; int i[8]; unsigned long long u[4]; };

#if __has_builtin(__builtin_amdgcn_global_load_async_to_lds_b128) && \
    __has_builtin(__builtin_amdgcn_s_wait_asynccnt)
#define USE_ASYNC_LDS 1
// Builtin signature (from hipcc diagnostics): 128-bit int-vector pointers,
// global (AS1) source first, LDS (AS3) destination, imm offset, imm cpol.
__device__ __forceinline__ void async_cp16(const signed char* g, signed char* l) {
  __builtin_amdgcn_global_load_async_to_lds_b128(
      (__attribute__((address_space(1))) v4i*)(__attribute__((address_space(1))) void*)g,
      (__attribute__((address_space(3))) v4i*)(__attribute__((address_space(3))) void*)l,
      /*offset=*/0, /*cpol=*/0);
}
#else
#define USE_ASYNC_LDS 0
#endif

// ---------------------------------------------------------------------------
// 8-bit WMMA fragment load: 16x64 tile, row-major, ld bytes between rows.
// Per ISA 7.12.2 (8-bit A 16x64 / B 64x16 layouts are the same pattern when
// the 64-length K vector is contiguous in memory):
//   lanes 0-15 : row = lane,    K chunks {0-7,16-23,32-39,48-55}
//   lanes 16-31: row = lane-16, K chunks {8-15,24-31,40-47,56-63}
// => four 8-byte loads per lane (global_load_b64 or ds_load_b64).
// ---------------------------------------------------------------------------
__device__ __forceinline__ v8i load_frag_i8(const signed char* tile, int ld) {
  int lane = threadIdx.x & 31;
  const signed char* p = tile + (long)(lane & 15) * ld + ((lane >> 4) << 3);
  vi8u f;
  f.u[0] = *(const unsigned long long*)(p);
  f.u[1] = *(const unsigned long long*)(p + 16);
  f.u[2] = *(const unsigned long long*)(p + 32);
  f.u[3] = *(const unsigned long long*)(p + 48);
  return f.v;
}

// ---------------------------------------------------------------------------
// K1: per-row symmetric int8 quant (per-token activations / per-out-channel
// weights). One wave per row, wave32 shuffle reduction.
// ---------------------------------------------------------------------------
__global__ __launch_bounds__(256) void quant_rows_kernel(
    const float* __restrict__ X, signed char* __restrict__ Q,
    float* __restrict__ SC, int rows, int K) {
  int wid = threadIdx.x >> 5, lane = threadIdx.x & 31;
  long row = (long)blockIdx.x * 8 + wid;
  if (row >= rows) return;
  const float* x = X + row * K;
  float amax = 0.f;
  for (int j = lane; j < K; j += 32) amax = fmaxf(amax, fabsf(x[j]));
  for (int m = 16; m >= 1; m >>= 1) amax = fmaxf(amax, __shfl_xor(amax, m, 32));
  float s = fmaxf(amax / QMAXF, 1e-8f);
  if (lane == 0) SC[row] = s;
  float inv = 1.f / s;
  signed char* q = Q + row * K;
  for (int j = lane; j < K; j += 32) {
    float v = fminf(fmaxf(rintf(x[j] * inv), -QMAXF), QMAXF);
    q[j] = (signed char)v;
  }
}

// ---------------------------------------------------------------------------
// K2: int8 GEMM  C[M,N] = (A_q[M,K] @ B_q[N,K]^T) * sa[m] * sb[n]
// Block = 256 threads = 8 waves, tile 64(M) x 128(N); each wave owns 32x32
// (2x2 WMMA accumulators), K swept in 64-wide IU8 WMMA steps.
// Async path: A(64x64) + B(128x64) tiles double-buffered in LDS via
// GLOBAL_LOAD_ASYNC_TO_LDS_B128 (3 x 16B per thread per step), ASYNCcnt
// waits overlap the next tile's DMA with this tile's 4 WMMAs.
// ---------------------------------------------------------------------------
__global__ __launch_bounds__(256) void gemm_i8_kernel(
    const signed char* __restrict__ A, const float* __restrict__ sa,
    const signed char* __restrict__ B, const float* __restrict__ sb,
    float* __restrict__ C, int M, int N, int K) {
  int tid = threadIdx.x;
  int wid = tid >> 5;
  int wm = wid & 1, wn = wid >> 1;
  long mblk = (long)blockIdx.x * 64;
  long nblk = (long)blockIdx.y * 128;
  long m0 = mblk + wm * 32;
  long n0 = nblk + wn * 32;

  vi8u acc[2][2] = {};

#if USE_ASYNC_LDS
  __shared__ signed char As[2][64][KSTEP];    // 2 x 4KB
  __shared__ signed char Bs[2][128][KSTEP];   // 2 x 8KB
  int lr = tid >> 2;              // 0..63
  int lc = (tid & 3) * 16;        // 0,16,32,48
  const int nsteps = K / KSTEP;

  auto stage = [&](int buf, int step) {
    long k0 = (long)step * KSTEP;
    async_cp16(A + (mblk + lr) * K + k0 + lc,      &As[buf][lr][lc]);
    async_cp16(B + (nblk + lr) * K + k0 + lc,      &Bs[buf][lr][lc]);
    async_cp16(B + (nblk + 64 + lr) * K + k0 + lc, &Bs[buf][64 + lr][lc]);
  };

  stage(0, 0);
  for (int step = 0; step < nsteps; ++step) {
    int buf = step & 1;
    if (step + 1 < nsteps) {
      stage(buf ^ 1, step + 1);                 // prefetch next slab (DMA)
      __builtin_amdgcn_s_wait_asynccnt(3);      // wait only for current slab
    } else {
      __builtin_amdgcn_s_wait_asynccnt(0);
    }
    __syncthreads();
    v8i a0 = load_frag_i8(&As[buf][wm * 32][0],      KSTEP);
    v8i a1 = load_frag_i8(&As[buf][wm * 32 + 16][0], KSTEP);
    v8i b0 = load_frag_i8(&Bs[buf][wn * 32][0],      KSTEP);
    v8i b1 = load_frag_i8(&Bs[buf][wn * 32 + 16][0], KSTEP);
    acc[0][0].v = __builtin_amdgcn_wmma_i32_16x16x64_iu8(true, a0, true, b0, acc[0][0].v, false, false);
    acc[0][1].v = __builtin_amdgcn_wmma_i32_16x16x64_iu8(true, a0, true, b1, acc[0][1].v, false, false);
    acc[1][0].v = __builtin_amdgcn_wmma_i32_16x16x64_iu8(true, a1, true, b0, acc[1][0].v, false, false);
    acc[1][1].v = __builtin_amdgcn_wmma_i32_16x16x64_iu8(true, a1, true, b1, acc[1][1].v, false, false);
    __syncthreads();                            // protect buf before re-stage
  }
#else
  for (int k0 = 0; k0 < K; k0 += KSTEP) {
    if (k0 + KSTEP < K) {  // L2 prefetch of next K-slab
      __builtin_prefetch(A + m0 * K + k0 + KSTEP, 0, 1);
      __builtin_prefetch(B + n0 * K + k0 + KSTEP, 0, 1);
    }
    v8i a0 = load_frag_i8(A + m0 * K + k0, K);
    v8i a1 = load_frag_i8(A + (m0 + 16) * K + k0, K);
    v8i b0 = load_frag_i8(B + n0 * K + k0, K);
    v8i b1 = load_frag_i8(B + (n0 + 16) * K + k0, K);
    acc[0][0].v = __builtin_amdgcn_wmma_i32_16x16x64_iu8(true, a0, true, b0, acc[0][0].v, false, false);
    acc[0][1].v = __builtin_amdgcn_wmma_i32_16x16x64_iu8(true, a0, true, b1, acc[0][1].v, false, false);
    acc[1][0].v = __builtin_amdgcn_wmma_i32_16x16x64_iu8(true, a1, true, b0, acc[1][0].v, false, false);
    acc[1][1].v = __builtin_amdgcn_wmma_i32_16x16x64_iu8(true, a1, true, b1, acc[1][1].v, false, false);
  }
#endif

  // C/D layout: VGPR r -> row = 8*half + r, col = lane&15 (within 16x16 tile)
  int lane = threadIdx.x & 31;
  int half = lane >> 4, col15 = lane & 15;
  for (int i = 0; i < 2; i++)
    for (int j = 0; j < 2; j++) {
      long n = n0 + j * 16 + col15;
      float sbn = sb[n];
      for (int r = 0; r < 8; r++) {
        long m = m0 + i * 16 + half * 8 + r;
        C[m * N + n] = (float)acc[i][j].i[r] * sa[m] * sbn;
      }
    }
}

// ---------------------------------------------------------------------------
// K3: fused RMSNorm + RoPE + per-(head,token) int8 quant for q / k.
// One wave per (token, head); lane L owns elements {L, L+32, L+64, L+96} so
// the rotate_half partner (j, j^64) stays inside the lane. Output head-major
// [NH, S, HD] int8 so attention fragments are contiguous-K loads.
// ---------------------------------------------------------------------------
__global__ __launch_bounds__(256) void norm_rope_quant_kernel(
    const float* __restrict__ X,        // [S, NH*HD]
    const float* __restrict__ w,        // [HD]
    const int*   __restrict__ pos,      // [S]
    signed char* __restrict__ Q,        // [NH, S, HD]
    float*       __restrict__ SC,       // [NH, S]
    int S, int NH) {
  int wid = threadIdx.x >> 5, lane = threadIdx.x & 31;
  int gw = blockIdx.x * 8 + wid;
  if (gw >= S * NH) return;
  int s = gw / NH, h = gw % NH;
  const float* x = X + (long)s * NH * HD + (long)h * HD;

  float x0 = x[lane], x1 = x[lane + 32], x2 = x[lane + 64], x3 = x[lane + 96];
  // RMSNorm (eps 1e-6)
  float ss = x0 * x0 + x1 * x1 + x2 * x2 + x3 * x3;
  for (int m = 16; m >= 1; m >>= 1) ss += __shfl_xor(ss, m, 32);
  float r = rsqrtf(ss * (1.f / (float)HD) + 1e-6f);
  x0 *= r * w[lane];      x1 *= r * w[lane + 32];
  x2 *= r * w[lane + 64]; x3 *= r * w[lane + 96];
  // RoPE: theta=1e6; inv_j = exp(-j * ln(1e6)/64), cos/sin index j mod 64.
  const float c = 13.815510558f / 64.f;
  float p = (float)pos[s];
  float f0 = p * __expf(-(float)lane * c);
  float f1 = p * __expf(-(float)(lane + 32) * c);
  float c0 = __cosf(f0), s0 = __sinf(f0), c1 = __cosf(f1), s1 = __sinf(f1);
  float o0 = x0 * c0 - x2 * s0;
  float o1 = x1 * c1 - x3 * s1;
  float o2 = x2 * c0 + x0 * s0;
  float o3 = x3 * c1 + x1 * s1;
  // per-(head,token) int8
  float amax = fmaxf(fmaxf(fabsf(o0), fabsf(o1)), fmaxf(fabsf(o2), fabsf(o3)));
  for (int m = 16; m >= 1; m >>= 1) amax = fmaxf(amax, __shfl_xor(amax, m, 32));
  float scale = fmaxf(amax / QMAXF, 1e-8f);
  if (lane == 0) SC[(long)h * S + s] = scale;
  float inv = 1.f / scale;
  signed char* q = Q + ((long)h * S + s) * HD;
  q[lane]      = (signed char)fminf(fmaxf(rintf(o0 * inv), -QMAXF), QMAXF);
  q[lane + 32] = (signed char)fminf(fmaxf(rintf(o1 * inv), -QMAXF), QMAXF);
  q[lane + 64] = (signed char)fminf(fmaxf(rintf(o2 * inv), -QMAXF), QMAXF);
  q[lane + 96] = (signed char)fminf(fmaxf(rintf(o3 * inv), -QMAXF), QMAXF);
}

// ---------------------------------------------------------------------------
// K4: v per-(head,token) int8 quant, stored TRANSPOSED [KV, HD, S] so that
// the PV WMMA B-fragment (column n = hd, K = keys) is a contiguous-K load.
// ---------------------------------------------------------------------------
__global__ __launch_bounds__(256) void v_quant_t_kernel(
    const float* __restrict__ X,        // [S, KV*HD]
    signed char* __restrict__ Q,        // [KV, HD, S]
    float*       __restrict__ SC,       // [KV, S]
    int S) {
  int wid = threadIdx.x >> 5, lane = threadIdx.x & 31;
  int gw = blockIdx.x * 8 + wid;
  if (gw >= S * KV_HEADS) return;
  int s = gw / KV_HEADS, h = gw % KV_HEADS;
  const float* x = X + (long)s * KV_HEADS * HD + (long)h * HD;
  float x0 = x[lane], x1 = x[lane + 32], x2 = x[lane + 64], x3 = x[lane + 96];
  float amax = fmaxf(fmaxf(fabsf(x0), fabsf(x1)), fmaxf(fabsf(x2), fabsf(x3)));
  for (int m = 16; m >= 1; m >>= 1) amax = fmaxf(amax, __shfl_xor(amax, m, 32));
  float scale = fmaxf(amax / QMAXF, 1e-8f);
  if (lane == 0) SC[(long)h * S + s] = scale;
  float inv = 1.f / scale;
  signed char* q = Q + (long)h * HD * S + s;   // column s of [HD, S]
  q[(long)(lane)      * S] = (signed char)fminf(fmaxf(rintf(x0 * inv), -QMAXF), QMAXF);
  q[(long)(lane + 32) * S] = (signed char)fminf(fmaxf(rintf(x1 * inv), -QMAXF), QMAXF);
  q[(long)(lane + 64) * S] = (signed char)fminf(fmaxf(rintf(x2 * inv), -QMAXF), QMAXF);
  q[(long)(lane + 96) * S] = (signed char)fminf(fmaxf(rintf(x3 * inv), -QMAXF), QMAXF);
}

// ---------------------------------------------------------------------------
// K5: causal flash attention, int8 WMMA for QK^T and PV.
// One wave per (head, 16-query tile). 64-key blocks. Score tile layout per
// ISA: row = 8*half + vgpr_idx, col = lane&15; softmax row stats via 16-lane
// xor-shuffle reductions. p is restaged through LDS (1KB/wave) to convert
// from C-layout to A-fragment layout (same-wave DS ops are in-order; the
// s_wait_dscnt + memory clobber pins ordering). Per-token v scales are
// folded into p before quantization (they live on the K dim of PV).
// Waves have divergent causal trip counts => no cross-wave barriers here;
// k/v fragments come straight from L1/L2 (waves in a block share one head).
// ---------------------------------------------------------------------------
__global__ __launch_bounds__(256) void attn_kernel(
    const signed char* __restrict__ Qi, const float* __restrict__ Qs,   // [H,S,HD], [H,S]
    const signed char* __restrict__ Ki, const float* __restrict__ Ks,   // [KV,S,HD], [KV,S]
    const signed char* __restrict__ Vt, const float* __restrict__ Vs,   // [KV,HD,S], [KV,S]
    float* __restrict__ O, int S) {                                     // [S, H*HD]
  __shared__ signed char p_lds[8][16][64];
  int wid = threadIdx.x >> 5, lane = threadIdx.x & 31;
  int half = lane >> 4, lane15 = lane & 15;
  int qtiles = S >> 4;
  int gw = blockIdx.x * 8 + wid;
  int h = gw / qtiles, qt = gw % qtiles;
  int kv = h / (H_HEADS / KV_HEADS);

  const signed char* qbase = Qi + ((long)h * S + qt * 16) * HD;
  v8i aq0 = load_frag_i8(qbase, HD);        // hd 0..63
  v8i aq1 = load_frag_i8(qbase + 64, HD);   // hd 64..127

  float qs[8];
  for (int r = 0; r < 8; r++) qs[r] = Qs[(long)h * S + qt * 16 + half * 8 + r];

  float mrow[8], lrow[8], o[8][8];          // o[hd_tile][row]
  for (int r = 0; r < 8; r++) {
    mrow[r] = -1e30f; lrow[r] = 0.f;
    for (int t = 0; t < 8; t++) o[t][r] = 0.f;
  }

  const float sfac = 0.08838834764831845f;  // HD^-0.5
  int nkt = (qt * 16 + 15) / 64 + 1;
  for (int kt = 0; kt < nkt; kt++) {
    // ---- scores: 4 x (16x16) via chained IU8 WMMAs over hd ----
    float sc[4][8];
    const signed char* kbase = Ki + ((long)kv * S + kt * 64) * HD;
    for (int t = 0; t < 4; t++) {
      v8i b0 = load_frag_i8(kbase + (long)(t * 16) * HD, HD);
      v8i b1 = load_frag_i8(kbase + (long)(t * 16) * HD + 64, HD);
      vi8u sa; v8i z = {}; sa.v = z;
      sa.v = __builtin_amdgcn_wmma_i32_16x16x64_iu8(true, aq0, true, b0, sa.v, false, false);
      sa.v = __builtin_amdgcn_wmma_i32_16x16x64_iu8(true, aq1, true, b1, sa.v, false, false);
      int col = kt * 64 + t * 16 + lane15;
      float ksc = Ks[(long)kv * S + col] * sfac;
      for (int r = 0; r < 8; r++) {
        int rowg = qt * 16 + half * 8 + r;
        float v = (float)sa.i[r] * qs[r] * ksc;
        sc[t][r] = (col > rowg) ? -1e9f : v;   // causal mask
      }
    }
    // ---- online softmax + fold per-token v scale into p ----
    float vsc[4];
    for (int t = 0; t < 4; t++) vsc[t] = Vs[(long)kv * S + kt * 64 + t * 16 + lane15];
    float pv[4][8], pmax[8];
    for (int r = 0; r < 8; r++) {
      float rm = fmaxf(fmaxf(sc[0][r], sc[1][r]), fmaxf(sc[2][r], sc[3][r]));
      for (int m = 8; m >= 1; m >>= 1) rm = fmaxf(rm, __shfl_xor(rm, m, 32));
      float mnew = fmaxf(mrow[r], rm);
      float corr = __expf(mrow[r] - mnew);
      mrow[r] = mnew;
      float rsum = 0.f, pm = 0.f;
      for (int t = 0; t < 4; t++) {
        float e = __expf(sc[t][r] - mnew);
        rsum += e;
        float pvv = e * vsc[t];
        pv[t][r] = pvv;
        pm = fmaxf(pm, pvv);
      }
      for (int m = 8; m >= 1; m >>= 1) {
        rsum += __shfl_xor(rsum, m, 32);
        pm = fmaxf(pm, __shfl_xor(pm, m, 32));
      }
      lrow[r] = lrow[r] * corr + rsum;
      pmax[r] = pm;
      for (int t = 0; t < 8; t++) o[t][r] *= corr;
    }
    // ---- quantize p (C-layout) into LDS as an A-layout 16x64 tile ----
    for (int r = 0; r < 8; r++) {
      float inv = QMAXF / fmaxf(pmax[r], 1e-20f);
      for (int t = 0; t < 4; t++) {
        int q = (int)rintf(pv[t][r] * inv);
        p_lds[wid][half * 8 + r][t * 16 + lane15] = (signed char)q;
      }
    }
    asm volatile("s_wait_dscnt 0" ::: "memory");
    v8i ap = load_frag_i8(&p_lds[wid][0][0], 64);
    // ---- PV: p (unsigned 0..127) x v_t, 8 hd tiles ----
    for (int t = 0; t < 8; t++) {
      const signed char* vb = Vt + ((long)kv * HD + t * 16) * S + kt * 64;
      v8i bv = load_frag_i8(vb, S);
      vi8u pa; v8i z = {}; pa.v = z;
      pa.v = __builtin_amdgcn_wmma_i32_16x16x64_iu8(false, ap, true, bv, pa.v, false, false);
      for (int r = 0; r < 8; r++)
        o[t][r] += (float)pa.i[r] * (pmax[r] * (1.f / QMAXF));
    }
  }
  // ---- normalize and write [S, H*HD] ----
  for (int t = 0; t < 8; t++)
    for (int r = 0; r < 8; r++) {
      long rowg = qt * 16 + half * 8 + r;
      long colg = (long)h * HD + t * 16 + lane15;
      O[rowg * (H_HEADS * HD) + colg] = o[t][r] / lrow[r];
    }
}

// ---------------------------------------------------------------------------
// Host-side launch. Inputs (setup_inputs order):
// 0 hidden_states [1,2048,2048] f32, 1 Wq [2048,2048], 2 Wk [1024,2048],
// 3 Wv [1024,2048], 4 Wo [2048,2048], 5 q_norm_w [128], 6 k_norm_w [128],
// 7 position_ids [1,2048] i32.  Output: [2048, 2048] f32.
// ---------------------------------------------------------------------------
extern "C" void kernel_launch(void* const* d_in, const int* in_sizes, int n_in,
                              void* d_out, int out_size, void* d_ws, size_t ws_size,
                              hipStream_t stream) {
  const int S = 2048, D = 2048;
  const int NQ = H_HEADS * HD;     // 2048
  const int NKV = KV_HEADS * HD;   // 1024

  const float* x   = (const float*)d_in[0];
  const float* Wq  = (const float*)d_in[1];
  const float* Wk  = (const float*)d_in[2];
  const float* Wv  = (const float*)d_in[3];
  const float* Wo  = (const float*)d_in[4];
  const float* qnw = (const float*)d_in[5];
  const float* knw = (const float*)d_in[6];
  const int*   pos = (const int*)d_in[7];
  float* out = (float*)d_out;

  // bump-allocate scratch (total ~80MB)
  char* ws = (char*)d_ws;
  size_t off = 0;
  auto alloc = [&](size_t bytes) -> char* {
    char* p = ws + off;
    off = (off + bytes + 255) & ~(size_t)255;
    return p;
  };
  signed char* x_i8    = (signed char*)alloc((size_t)S * D);
  signed char* wq_i8   = (signed char*)alloc((size_t)NQ * D);
  signed char* wk_i8   = (signed char*)alloc((size_t)NKV * D);
  signed char* wv_i8   = (signed char*)alloc((size_t)NKV * D);
  signed char* wo_i8   = (signed char*)alloc((size_t)D * NQ);
  signed char* at_i8   = (signed char*)alloc((size_t)S * NQ);
  float* x_s  = (float*)alloc((size_t)S * 4);
  float* wq_s = (float*)alloc((size_t)NQ * 4);
  float* wk_s = (float*)alloc((size_t)NKV * 4);
  float* wv_s = (float*)alloc((size_t)NKV * 4);
  float* wo_s = (float*)alloc((size_t)D * 4);
  float* at_s = (float*)alloc((size_t)S * 4);
  float* qf = (float*)alloc((size_t)S * NQ * 4);
  float* kf = (float*)alloc((size_t)S * NKV * 4);
  float* vf = (float*)alloc((size_t)S * NKV * 4);
  signed char* q_i8 = (signed char*)alloc((size_t)H_HEADS * S * HD);
  signed char* k_i8 = (signed char*)alloc((size_t)KV_HEADS * S * HD);
  signed char* vt_i8 = (signed char*)alloc((size_t)KV_HEADS * HD * S);
  float* q_s = (float*)alloc((size_t)H_HEADS * S * 4);
  float* k_s = (float*)alloc((size_t)KV_HEADS * S * 4);
  float* v_s = (float*)alloc((size_t)KV_HEADS * S * 4);
  float* attn_f = (float*)alloc((size_t)S * NQ * 4);
  (void)ws_size; (void)in_sizes; (void)n_in; (void)out_size;

  // 1) int8 quantization of activations + weights
  quant_rows_kernel<<<S / 8, 256, 0, stream>>>(x, x_i8, x_s, S, D);
  quant_rows_kernel<<<NQ / 8, 256, 0, stream>>>(Wq, wq_i8, wq_s, NQ, D);
  quant_rows_kernel<<<NKV / 8, 256, 0, stream>>>(Wk, wk_i8, wk_s, NKV, D);
  quant_rows_kernel<<<NKV / 8, 256, 0, stream>>>(Wv, wv_i8, wv_s, NKV, D);
  quant_rows_kernel<<<D / 8, 256, 0, stream>>>(Wo, wo_i8, wo_s, D, D);

  // 2) Q/K/V projections (int8 WMMA GEMM, async-LDS double buffered)
  gemm_i8_kernel<<<dim3(S / 64, NQ / 128), 256, 0, stream>>>(
      x_i8, x_s, wq_i8, wq_s, qf, S, NQ, D);
  gemm_i8_kernel<<<dim3(S / 64, NKV / 128), 256, 0, stream>>>(
      x_i8, x_s, wk_i8, wk_s, kf, S, NKV, D);
  gemm_i8_kernel<<<dim3(S / 64, NKV / 128), 256, 0, stream>>>(
      x_i8, x_s, wv_i8, wv_s, vf, S, NKV, D);

  // 3) RMSNorm + RoPE + per-(head,token) quant; v quant-transpose
  norm_rope_quant_kernel<<<(S * H_HEADS) / 8, 256, 0, stream>>>(
      qf, qnw, pos, q_i8, q_s, S, H_HEADS);
  norm_rope_quant_kernel<<<(S * KV_HEADS) / 8, 256, 0, stream>>>(
      kf, knw, pos, k_i8, k_s, S, KV_HEADS);
  v_quant_t_kernel<<<(S * KV_HEADS) / 8, 256, 0, stream>>>(vf, vt_i8, v_s, S);

  // 4) causal flash attention (int8 WMMA QK^T and PV)
  attn_kernel<<<(H_HEADS * (S / 16)) / 8, 256, 0, stream>>>(
      q_i8, q_s, k_i8, k_s, vt_i8, v_s, attn_f, S);

  // 5) output projection
  quant_rows_kernel<<<S / 8, 256, 0, stream>>>(attn_f, at_i8, at_s, S, NQ);
  gemm_i8_kernel<<<dim3(S / 64, D / 128), 256, 0, stream>>>(
      at_i8, at_s, wo_i8, wo_s, out, S, D, NQ);
}
